// GPT_43069932044441
// MI455X (gfx1250) — matrix operand order repaired and checked by reading, not compiled, not run on hardware
//
#include <hip/hip_runtime.h>
#include <math.h>

typedef __attribute__((ext_vector_type(16))) __bf16 v16bf;
typedef __attribute__((ext_vector_type(8)))  __bf16 v8bf;
typedef __attribute__((ext_vector_type(8)))  float  v8f;
typedef __attribute__((ext_vector_type(4)))  float  v4f;
typedef __attribute__((ext_vector_type(4)))  unsigned int u32x4;
typedef __attribute__((ext_vector_type(8)))  int    i32x8;
typedef __attribute__((ext_vector_type(4)))  int    i32x4;

#define NTHREADS 256

// ---------------------------------------------------------------------------
// TDM 2-D tile load: global bf16 [rows x 32] (row pitch strideElems) -> LDS.
// D# per CDNA5 ISA. Uniform operands; wave-level.
// This toolchain exposes the 6-arg builtin:
//   (u32x4 g0, i32x8 g1, i32x4 g2, i32x4 g3, i32x8 aux, i32 cpol)
// ---------------------------------------------------------------------------
__device__ __forceinline__ void tdm_load_2d(const __bf16* gptr, unsigned lds_addr,
                                            int tileRows, unsigned strideElems)
{
    unsigned long long ga = (unsigned long long)(size_t)gptr;
    u32x4 g0;
    g0[0] = 1u;                                       // count=1, user descriptor
    g0[1] = lds_addr;                                 // LDS byte address
    g0[2] = (unsigned)(ga & 0xFFFFFFFFu);             // global_addr[31:0]
    g0[3] = (unsigned)((ga >> 32) & 0x01FFFFFFu)      // global_addr[56:32]
          | 0x80000000u;                              // type=2 (bits 127:126 = 10b)
    unsigned td0 = 32u;                               // tensor_dim0 (elems/row)
    unsigned td1 = (unsigned)tileRows;                // tensor_dim1 (rows)
    i32x8 g1;
    g1[0] = 0x00010000;                               // wg_mask=0, data_size=1 (2B)
    g1[1] = (int)((td0 & 0xFFFFu) << 16);             // tensor_dim0 @ bit48
    g1[2] = (int)((td0 >> 16) | ((td1 & 0xFFFFu) << 16));
    g1[3] = (int)((td1 >> 16) | (32u << 16));         // tile_dim0 = 32
    g1[4] = (int)(td1 & 0xFFFFu);                     // tile_dim1 = rows, tile_dim2=0
    g1[5] = (int)strideElems;                         // tensor_dim0_stride[31:0]
    g1[6] = 0;
    g1[7] = 0;
    i32x4 z4 = {0, 0, 0, 0};
    i32x8 z8 = {0, 0, 0, 0, 0, 0, 0, 0};
    __builtin_amdgcn_tensor_load_to_lds(g0, g1, z4, z4, z8, 0);
}

// ---------------------------------------------------------------------------
// Embedding: x[b,t,c] = wte[idx[b,t]][c] + wpe[t][c]   (f32 residual stream)
// ---------------------------------------------------------------------------
__global__ __launch_bounds__(NTHREADS) void embed_kernel(
    const int* __restrict__ idx, const float* __restrict__ wte,
    const float* __restrict__ wpe, float* __restrict__ x,
    int T, int C, long long n)
{
    long long i = (long long)blockIdx.x * blockDim.x + threadIdx.x;
    if (i >= n) return;
    int c = (int)(i % C);
    long long bt = i / C;
    int t = (int)(bt % T);
    int tok = idx[bt];
    x[i] = wte[(size_t)tok * C + c] + wpe[(size_t)t * C + c];
}

// ---------------------------------------------------------------------------
// f32 -> bf16 (same layout)
// ---------------------------------------------------------------------------
__global__ __launch_bounds__(NTHREADS) void cvt_kernel(
    const float* __restrict__ in, __bf16* __restrict__ out, long long n)
{
    long long i = (long long)blockIdx.x * blockDim.x + threadIdx.x;
    if (i < n) out[i] = (__bf16)in[i];
}

// ---------------------------------------------------------------------------
// f32 [K,N] -> bf16 [N,K] (transpose + convert, for weight matrices)
// ---------------------------------------------------------------------------
__global__ __launch_bounds__(NTHREADS) void transpose_cvt_kernel(
    const float* __restrict__ in, __bf16* __restrict__ out, int K, int N)
{
    long long i = (long long)blockIdx.x * blockDim.x + threadIdx.x;
    if (i >= (long long)K * N) return;
    int k = (int)(i / N), n = (int)(i % N);
    out[(size_t)n * K + k] = (__bf16)in[i];
}

// ---------------------------------------------------------------------------
// V transpose: qkv16[b,t, 2C + h*D + d] -> vtb[b,h,d,t]   (bf16)
// ---------------------------------------------------------------------------
__global__ __launch_bounds__(NTHREADS) void vtrans_kernel(
    const __bf16* __restrict__ qkv16, __bf16* __restrict__ vtb,
    int T, int H, int Dh, int C3, long long n)
{
    long long i = (long long)blockIdx.x * blockDim.x + threadIdx.x;
    if (i >= n) return;
    int t = (int)(i % T);
    int d = (int)((i / T) % Dh);
    int h = (int)((i / ((long long)T * Dh)) % H);
    long long b = i / ((long long)T * Dh * H);
    vtb[i] = qkv16[(size_t)(b * T + t) * C3 + 2 * 768 + h * Dh + d];
}

// ---------------------------------------------------------------------------
// LayerNorm: one block per row; writes f32 and/or bf16 outputs
// ---------------------------------------------------------------------------
__global__ __launch_bounds__(NTHREADS) void ln_kernel(
    const float* __restrict__ in, const float* __restrict__ g,
    const float* __restrict__ b, float* __restrict__ outf,
    __bf16* __restrict__ out16, int C)
{
    int row = blockIdx.x;
    const float* x = in + (size_t)row * C;
    __shared__ float red[NTHREADS];

    float s = 0.f;
    for (int c = threadIdx.x; c < C; c += NTHREADS) s += x[c];
    red[threadIdx.x] = s; __syncthreads();
    for (int k = NTHREADS / 2; k > 0; k >>= 1) {
        if (threadIdx.x < k) red[threadIdx.x] += red[threadIdx.x + k];
        __syncthreads();
    }
    float mean = red[0] / C;
    __syncthreads();

    float v = 0.f;
    for (int c = threadIdx.x; c < C; c += NTHREADS) {
        float d = x[c] - mean; v += d * d;
    }
    red[threadIdx.x] = v; __syncthreads();
    for (int k = NTHREADS / 2; k > 0; k >>= 1) {
        if (threadIdx.x < k) red[threadIdx.x] += red[threadIdx.x + k];
        __syncthreads();
    }
    float rstd = rsqrtf(red[0] / C + 1e-5f);
    for (int c = threadIdx.x; c < C; c += NTHREADS) {
        float o = (x[c] - mean) * rstd * g[c] + b[c];
        if (outf)  outf[(size_t)row * C + c]  = o;
        if (out16) out16[(size_t)row * C + c] = (__bf16)o;
    }
}

// ---------------------------------------------------------------------------
// WMMA GEMM, TDM-staged, double-buffered LDS.
//   D = act(scale * A@B^T + bias [+ resid])
//   A: bf16 [M,K] rows (pitch lda);  B: bf16 [N,K] rows (pitch ldb)
//   Block tile 128x64x32; 8 wave32 in 4(M) x 2(N); wave tile 32x32 (4 WMMA).
//   Tiles DMA'd by TENSOR_LOAD_TO_LDS (wave 0), synced via TENSORcnt+barrier.
// ---------------------------------------------------------------------------
template <int ACT, bool OUT16>   // ACT: 0 none, 1 exact GELU
__global__ __launch_bounds__(NTHREADS) void gemm_wmma(
    const __bf16* __restrict__ Abase, int lda, long long sAo, long long sAi,
    const __bf16* __restrict__ Bbase, int ldb, long long sBo, long long sBi,
    void* __restrict__ Dbase, int ldd, long long sDo, long long sDi,
    const float* __restrict__ bias, const float* __restrict__ resid,
    int M, int N, int K, int Hinner, float scale)
{
    __shared__ __align__(64) __bf16 As[2][128 * 32];
    __shared__ __align__(64) __bf16 Bs[2][64 * 32];

    int z  = blockIdx.z;
    int zo = z / Hinner, zi = z % Hinner;
    const __bf16* A = Abase + zo * sAo + zi * sAi;
    const __bf16* B = Bbase + zo * sBo + zi * sBi;

    const int tid    = threadIdx.x;
    const int w      = tid >> 5;
    const int lane   = tid & 31;
    const int half   = lane >> 4;
    const int r      = lane & 15;
    const int waveM  = w & 3;
    const int waveN  = w >> 2;
    const int blockM = blockIdx.y * 128;
    const int blockN = blockIdx.x * 64;

    auto stage = [&](int kk, int buf) {     // wave 0 issues both tile DMAs
        if (tid == 0) {
            tdm_load_2d(A + (size_t)blockM * lda + kk,
                        (unsigned)(size_t)&As[buf][0], 128, (unsigned)lda);
            tdm_load_2d(B + (size_t)blockN * ldb + kk,
                        (unsigned)(size_t)&Bs[buf][0], 64, (unsigned)ldb);
        }
    };

    v8f acc[2][2] = {};
    int buf = 0;
    stage(0, 0);
    if (tid < 32) __builtin_amdgcn_s_wait_tensorcnt((short)0);
    __syncthreads();

    for (int kk = 0; kk < K; kk += 32) {
        if (kk + 32 < K) stage(kk + 32, buf ^ 1);

        // A fragments: row = waveM*32 + mi*16 + r;
        //   elem j -> k = half*16 + j  => one 32B chunk
        v16bf afrag[2];
        #pragma unroll
        for (int mi = 0; mi < 2; ++mi) {
            const __bf16* arow = &As[buf][(waveM * 32 + mi * 16 + r) * 32];
            afrag[mi] = *(const v16bf*)(arow + half * 16);
        }
        // B fragments: col = waveN*32 + ni*16 + r; elem j -> k = half*16 + j
        v16bf bfrag[2];
        #pragma unroll
        for (int ni = 0; ni < 2; ++ni) {
            const __bf16* brow = &Bs[buf][(waveN * 32 + ni * 16 + r) * 32];
            bfrag[ni] = *(const v16bf*)(brow + half * 16);
        }
        #pragma unroll
        for (int mi = 0; mi < 2; ++mi)
            #pragma unroll
            for (int ni = 0; ni < 2; ++ni)
                acc[mi][ni] = __builtin_amdgcn_wmma_f32_16x16x32_bf16(
                    false, afrag[mi], false, bfrag[ni],
                    (short)0, acc[mi][ni], false, false);

        if (tid < 32) __builtin_amdgcn_s_wait_tensorcnt((short)0);
        __syncthreads();
        buf ^= 1;
    }

    // Epilogue: C/D layout VGPR g -> M = subtileM + half*8 + g, N = subtileN + r
    #pragma unroll
    for (int mi = 0; mi < 2; ++mi) {
        #pragma unroll
        for (int ni = 0; ni < 2; ++ni) {
            int n = blockN + waveN * 32 + ni * 16 + r;
            float bv = bias ? bias[n] : 0.0f;
            #pragma unroll
            for (int g = 0; g < 8; ++g) {
                int m = blockM + waveM * 32 + mi * 16 + (half << 3) + g;
                float v = acc[mi][ni][g] * scale + bv;
                if (ACT == 1) v = 0.5f * v * (1.0f + erff(v * 0.70710678118654752f));
                size_t di = (size_t)m * ldd + n;
                if (OUT16) {
                    ((__bf16*)Dbase + zo * sDo + zi * sDi)[di] = (__bf16)v;
                } else {
                    float* Dp = (float*)Dbase + zo * sDo + zi * sDi;
                    if (resid) v += (resid + zo * sDo + zi * sDi)[di];
                    Dp[di] = v;
                }
            }
        }
    }
}

// ---------------------------------------------------------------------------
// Causal softmax: read f32 scores, write bf16 probabilities
// ---------------------------------------------------------------------------
__global__ __launch_bounds__(NTHREADS) void softmax_causal_kernel(
    const float* __restrict__ S, __bf16* __restrict__ P, int T)
{
    long long row = blockIdx.x;
    int t = (int)(row % T);
    const float* p = S + row * (long long)T;
    __bf16* q = P + row * (long long)T;
    __shared__ float red[NTHREADS];

    float mx = -INFINITY;
    for (int s = threadIdx.x; s < T; s += NTHREADS)
        if (s <= t) mx = fmaxf(mx, p[s]);
    red[threadIdx.x] = mx; __syncthreads();
    for (int k = NTHREADS / 2; k > 0; k >>= 1) {
        if (threadIdx.x < k) red[threadIdx.x] = fmaxf(red[threadIdx.x], red[threadIdx.x + k]);
        __syncthreads();
    }
    mx = red[0]; __syncthreads();

    float sum = 0.f;
    for (int s = threadIdx.x; s < T; s += NTHREADS)
        if (s <= t) sum += expf(p[s] - mx);
    red[threadIdx.x] = sum; __syncthreads();
    for (int k = NTHREADS / 2; k > 0; k >>= 1) {
        if (threadIdx.x < k) red[threadIdx.x] += red[threadIdx.x + k];
        __syncthreads();
    }
    float inv = 1.0f / red[0];

    for (int s = threadIdx.x; s < T; s += NTHREADS)
        q[s] = (__bf16)((s <= t) ? expf(p[s] - mx) * inv : 0.0f);
}

// ---------------------------------------------------------------------------
// NLL loss: one block per token row of logits[V]; ignore_index == 0
// ---------------------------------------------------------------------------
__global__ __launch_bounds__(NTHREADS) void loss_kernel(
    const float* __restrict__ logits, const int* __restrict__ targets,
    float* __restrict__ acc, int V)
{
    int row = blockIdx.x;
    const float* p = logits + (size_t)row * V;
    __shared__ float red[NTHREADS];

    float mx = -INFINITY;
    for (int i = threadIdx.x; i < V; i += NTHREADS) mx = fmaxf(mx, p[i]);
    red[threadIdx.x] = mx; __syncthreads();
    for (int k = NTHREADS / 2; k > 0; k >>= 1) {
        if (threadIdx.x < k) red[threadIdx.x] = fmaxf(red[threadIdx.x], red[threadIdx.x + k]);
        __syncthreads();
    }
    mx = red[0]; __syncthreads();

    float sum = 0.f;
    for (int i = threadIdx.x; i < V; i += NTHREADS) sum += expf(p[i] - mx);
    red[threadIdx.x] = sum; __syncthreads();
    for (int k = NTHREADS / 2; k > 0; k >>= 1) {
        if (threadIdx.x < k) red[threadIdx.x] += red[threadIdx.x + k];
        __syncthreads();
    }

    if (threadIdx.x == 0) {
        int tgt = targets[row];
        if (tgt != 0) {
            float nll = -(p[tgt] - mx - logf(red[0]));
            atomicAdd(&acc[0], nll);
            atomicAdd(&acc[1], 1.0f);
        }
    }
}

__global__ void finalize_loss_kernel(const float* __restrict__ acc,
                                     float* __restrict__ loss_out)
{
    loss_out[0] = acc[0] / fmaxf(acc[1], 1.0f);
}

// ---------------------------------------------------------------------------
// Host orchestration
// ---------------------------------------------------------------------------
extern "C" void kernel_launch(void* const* d_in, const int* in_sizes, int n_in,
                              void* d_out, int out_size, void* d_ws, size_t ws_size,
                              hipStream_t stream)
{
    (void)in_sizes; (void)n_in; (void)out_size; (void)ws_size;

    const int L = 6, C = 768, H = 12, V = 32000, B = 4, T = 1024, Dh = 64;
    const int M = B * T;
    const int C3 = 3 * C, C4 = 4 * C;

    const int*   idx     = (const int*)  d_in[0];
    const int*   targets = (const int*)  d_in[1];
    const float* wte     = (const float*)d_in[2];
    const float* wpe     = (const float*)d_in[3];
    const float* ln1_g   = (const float*)d_in[4];
    const float* ln1_b   = (const float*)d_in[5];
    const float* attn_w  = (const float*)d_in[6];
    const float* attn_b  = (const float*)d_in[7];
    const float* proj_w  = (const float*)d_in[8];
    const float* proj_b  = (const float*)d_in[9];
    const float* ln2_g   = (const float*)d_in[10];
    const float* ln2_b   = (const float*)d_in[11];
    const float* fc_w    = (const float*)d_in[12];
    const float* fc_b    = (const float*)d_in[13];
    const float* fc2_w   = (const float*)d_in[14];
    const float* fc2_b   = (const float*)d_in[15];
    const float* lnf_g   = (const float*)d_in[16];
    const float* lnf_b   = (const float*)d_in[17];

    // d_out layout: logits [M*V] | loss [1] | x [M*C]
    float* out_logits = (float*)d_out;
    float* out_loss   = out_logits + (size_t)M * V;
    float* out_x      = out_loss + 1;

    // Workspace carve-up (byte-based, 256B aligned chunks)
    char*  wsb = (char*)d_ws;
    size_t off = 0;
    auto carveB = [&](size_t nbytes) {
        void* p = (void*)(wsb + off);
        off += (nbytes + 255) & ~(size_t)255;
        return p;
    };
    float*  xbuf  = (float*) carveB((size_t)M * C * 4);          // residual (f32)
    __bf16* h16   = (__bf16*)carveB((size_t)M * C * 2);          // LN out
    __bf16* qkv16 = (__bf16*)carveB((size_t)M * C3 * 2);         // qkv
    __bf16* y16   = (__bf16*)carveB((size_t)M * C * 2);          // attn out
    __bf16* m16   = (__bf16*)carveB((size_t)M * C4 * 2);         // MLP hidden
    __bf16* lnf16 = (__bf16*)carveB((size_t)M * C * 2);          // final LN (bf16)
    __bf16* vtb   = (__bf16*)carveB((size_t)M * C * 2);          // V^T per head
    float*  sbuf  = (float*) carveB((size_t)B * H * T * T * 4);  // scores f32
    __bf16* pb16  = (__bf16*)carveB((size_t)B * H * T * T * 2);  // probs bf16
    __bf16* awT   = (__bf16*)carveB((size_t)L * C3 * C * 2);     // [2304,768] x L
    __bf16* pwT   = (__bf16*)carveB((size_t)L * C * C * 2);      // [768,768] x L
    __bf16* fwT   = (__bf16*)carveB((size_t)L * C4 * C * 2);     // [3072,768] x L
    __bf16* f2wT  = (__bf16*)carveB((size_t)L * C * C4 * 2);     // [768,3072] x L
    __bf16* wte16 = (__bf16*)carveB((size_t)V * C * 2);          // [32000,768]
    float*  accb  = (float*) carveB(2 * 4);

    auto grid1d = [](long long n) { return (int)((n + NTHREADS - 1) / NTHREADS); };

    // --- one-time (per launch) weight conversion: f32 -> bf16, B as [N,K] ---
    for (int l = 0; l < L; ++l) {
        transpose_cvt_kernel<<<grid1d((long long)C * C3), NTHREADS, 0, stream>>>(
            attn_w + (size_t)l * C * C3, awT + (size_t)l * C3 * C, C, C3);
        transpose_cvt_kernel<<<grid1d((long long)C * C), NTHREADS, 0, stream>>>(
            proj_w + (size_t)l * C * C, pwT + (size_t)l * C * C, C, C);
        transpose_cvt_kernel<<<grid1d((long long)C * C4), NTHREADS, 0, stream>>>(
            fc_w + (size_t)l * C * C4, fwT + (size_t)l * C4 * C, C, C4);
        transpose_cvt_kernel<<<grid1d((long long)C4 * C), NTHREADS, 0, stream>>>(
            fc2_w + (size_t)l * C4 * C, f2wT + (size_t)l * C * C4, C4, C);
    }
    cvt_kernel<<<grid1d((long long)V * C), NTHREADS, 0, stream>>>(
        wte, wte16, (long long)V * C);   // wte already [N,K] for the LM head

    // --- embedding ---
    embed_kernel<<<grid1d((long long)M * C), NTHREADS, 0, stream>>>(
        idx, wte, wpe, xbuf, T, C, (long long)M * C);

    const long long sQKVo = (long long)T * C3;
    const long long sSo   = (long long)H * T * T;
    const long long sSi   = (long long)T * T;
    const long long sYo   = (long long)T * C;
    const long long sVo   = (long long)H * Dh * T;
    const long long sVi   = (long long)Dh * T;

    for (int l = 0; l < L; ++l) {
        const float*  ab  = attn_b + (size_t)l * C3;
        const float*  pb  = proj_b + (size_t)l * C;
        const float*  fb  = fc_b   + (size_t)l * C4;
        const float*  f2b = fc2_b  + (size_t)l * C;
        const __bf16* aw  = awT  + (size_t)l * C3 * C;
        const __bf16* pw  = pwT  + (size_t)l * C * C;
        const __bf16* fw  = fwT  + (size_t)l * C4 * C;
        const __bf16* f2w = f2wT + (size_t)l * C * C4;

        // h16 = ln1(x)
        ln_kernel<<<M, NTHREADS, 0, stream>>>(xbuf, ln1_g + (size_t)l * C,
                                              ln1_b + (size_t)l * C, nullptr, h16, C);

        // qkv16 = h @ aw^T + ab                     [4096 x 2304 x 768] -> bf16
        gemm_wmma<0, true><<<dim3(C3 / 64, M / 128, 1), NTHREADS, 0, stream>>>(
            h16, C, 0, 0,  aw, C, 0, 0,  qkv16, C3, 0, 0,
            ab, nullptr, M, C3, C, 1, 1.0f);

        // vtb = V^T (per head)
        vtrans_kernel<<<grid1d((long long)M * C), NTHREADS, 0, stream>>>(
            qkv16, vtb, T, H, Dh, C3, (long long)M * C);

        // S = scale * Q @ K^T (batched B*H)         [1024 x 1024 x 64] -> f32
        gemm_wmma<0, false><<<dim3(T / 64, T / 128, B * H), NTHREADS, 0, stream>>>(
            qkv16,     C3, sQKVo, 64,
            qkv16 + C, C3, sQKVo, 64,
            sbuf,      T,  sSo,   sSi,
            nullptr, nullptr, T, T, Dh, H, 0.125f);

        // causal softmax -> bf16 probs
        softmax_causal_kernel<<<B * H * T, NTHREADS, 0, stream>>>(sbuf, pb16, T);

        // y16 = P @ V (batched)                     [1024 x 64 x 1024] -> bf16
        gemm_wmma<0, true><<<dim3(Dh / 64, T / 128, B * H), NTHREADS, 0, stream>>>(
            pb16, T, sSo, sSi,
            vtb,  T, sVo, sVi,
            y16,  C, sYo, 64,
            nullptr, nullptr, T, Dh, T, H, 1.0f);

        // x = x + y @ pw^T + pb                     [4096 x 768 x 768] -> f32
        gemm_wmma<0, false><<<dim3(C / 64, M / 128, 1), NTHREADS, 0, stream>>>(
            y16, C, 0, 0,  pw, C, 0, 0,  xbuf, C, 0, 0,
            pb, xbuf, M, C, C, 1, 1.0f);

        // h16 = ln2(x)
        ln_kernel<<<M, NTHREADS, 0, stream>>>(xbuf, ln2_g + (size_t)l * C,
                                              ln2_b + (size_t)l * C, nullptr, h16, C);

        // m16 = gelu(h @ fw^T + fb)                 [4096 x 3072 x 768] -> bf16
        gemm_wmma<1, true><<<dim3(C4 / 64, M / 128, 1), NTHREADS, 0, stream>>>(
            h16, C, 0, 0,  fw, C, 0, 0,  m16, C4, 0, 0,
            fb, nullptr, M, C4, C, 1, 1.0f);

        // x = x + m @ f2w^T + f2b                   [4096 x 768 x 3072] -> f32
        gemm_wmma<0, false><<<dim3(C / 64, M / 128, 1), NTHREADS, 0, stream>>>(
            m16, C4, 0, 0,  f2w, C4, 0, 0,  xbuf, C, 0, 0,
            f2b, xbuf, M, C, C4, 1, 1.0f);
    }

    // final LN -> f32 into d_out x slot, bf16 mirror for LM head
    ln_kernel<<<M, NTHREADS, 0, stream>>>(xbuf, lnf_g, lnf_b, out_x, lnf16, C);

    // logits = x @ wte^T                            [4096 x 32000 x 768] -> f32
    gemm_wmma<0, false><<<dim3(V / 64, M / 128, 1), NTHREADS, 0, stream>>>(
        lnf16, C, 0, 0,  wte16, C, 0, 0,  out_logits, V, 0, 0,
        nullptr, nullptr, M, V, C, 1, 1.0f);

    // loss
    (void)hipMemsetAsync(accb, 0, 2 * sizeof(float), stream);
    loss_kernel<<<M, NTHREADS, 0, stream>>>(out_logits, targets, accb, V);
    finalize_loss_kernel<<<1, 1, 0, stream>>>(accb, out_loss);
}
